// BahdanauAttention_84293028151683
// MI455X (gfx1250) — matrix operand order repaired and checked by reading, not compiled, not run on hardware
//
#include <hip/hip_runtime.h>
#include <hip/hip_bf16.h>
#include <cmath>

// Problem dims (fixed by the reference)
#define B_  32
#define T_  2048
#define D_  1024
#define U_  1024

typedef __attribute__((ext_vector_type(16))) __bf16 v16bf;
typedef __attribute__((ext_vector_type(8)))  float  v8f;
typedef int v4i __attribute__((vector_size(4 * sizeof(int))));  // builtin param type

#define LDS_ROW_STRIDE 1028   // 1024 + 4 pad floats: stride%64 == 4 -> conflict-free b128 reads

// ---------------------------------------------------------------------------
// Kernel 1: pack W1 (f32, [D,U] row-major) into bf16 B-fragment layout.
// B matrix for v_wmma_f32_16x16x32_bf16 is 32(K) x 16(N):
//   lane L (n = L&15, g = L>>4) holds column N = n, K = (e>>3)*16 + g*8 + (e&7)  (e = 0..15)
// Packed layout: pw[(((ct*32 + kc)*32) + lane)*16 + e]
// ---------------------------------------------------------------------------
__global__ __launch_bounds__(256) void pack_w1_kernel(const float* __restrict__ W1,
                                                      __bf16* __restrict__ pw) {
    int idx  = blockIdx.x * 256 + threadIdx.x;      // 0 .. 1048575
    int e    = idx & 15;
    int lane = (idx >> 4) & 31;
    int kc   = (idx >> 9) & 31;
    int ct   = idx >> 14;
    int K = kc * 32 + (e >> 3) * 16 + (lane >> 4) * 8 + (e & 7);
    int N = ct * 16 + (lane & 15);
    pw[idx] = (__bf16)W1[K * U_ + N];
}

// ---------------------------------------------------------------------------
// Kernel 2: proj_q[b,u] = query[b,:] @ W2[:,u] + b2[u]   (tiny, scalar)
// ---------------------------------------------------------------------------
__global__ __launch_bounds__(256) void proj_q_kernel(const float* __restrict__ q,
                                                     const float* __restrict__ W2,
                                                     const float* __restrict__ b2,
                                                     float* __restrict__ pq) {
    int idx = blockIdx.x * 256 + threadIdx.x;       // 0 .. 32767
    int b = idx >> 10, u = idx & 1023;
    const float* qr = q + b * D_;
    float acc = b2[u];
    #pragma unroll 4
    for (int d = 0; d < D_; ++d) acc += qr[d] * W2[d * U_ + u];
    pq[idx] = acc;
}

// ---------------------------------------------------------------------------
// Kernel 3 (hot): fused  score[b,t] = tanh(values@W1 + b1 + proj_q) @ V + bV
//                 + mask*(-1e9)
// One workgroup = (b, 16-row t-tile). The 16x1024 f32 values tile is staged
// in LDS once via async-to-LDS (8x L2-traffic reduction vs per-wave loads).
// 8 waves; wave w owns U columns [w*128, w*128+128) as 8 WMMA column tiles.
// A fragment (16-bit A 16x32, ISA 7.12.2): lane L (m=L&15, g=L>>4), elem e:
//   K = (e>=8)*16 + g*8 + (e&7)  -> two contiguous 8-float runs per lane.
// C/D layout: VGPR r, lane L: row M = r + g*8, col N = ct*16 + (L&15)
// ---------------------------------------------------------------------------
__global__ __launch_bounds__(256) void score_kernel(const float*  __restrict__ values,
                                                    const __bf16* __restrict__ pw,
                                                    const float*  __restrict__ pq,
                                                    const float*  __restrict__ b1,
                                                    const float*  __restrict__ Vv,
                                                    const float*  __restrict__ bV,
                                                    const float*  __restrict__ mask,
                                                    float* __restrict__ scores) {
    const int blk  = blockIdx.x;          // 0 .. 4095
    const int b    = blk >> 7;            // T_/16 = 128 tiles per batch
    const int t0   = (blk & 127) * 16;
    const int tid  = threadIdx.x;
    const int wave = tid >> 5;
    const int lane = tid & 31;
    const int g    = lane >> 4;
    const int n    = lane & 15;

    __shared__ float vtile[16 * LDS_ROW_STRIDE];    // 16 rows, padded
    __shared__ float rowsum[16];
    if (tid < 16) rowsum[tid] = 0.0f;

    // ---- stage the 16x1024 values tile into LDS (async copy, 16B/lane/iter)
    // iteration i: all 256 threads copy row i (4 KB contiguous both sides)
    #pragma unroll
    for (int i = 0; i < 16; ++i) {
        const float* gsrc = values + ((size_t)b * T_ + (t0 + i)) * (size_t)D_ + tid * 4;
        float*       ldst = &vtile[i * LDS_ROW_STRIDE + tid * 4];
#if __has_builtin(__builtin_amdgcn_global_load_async_to_lds_b128)
        __builtin_amdgcn_global_load_async_to_lds_b128((v4i*)gsrc, (v4i*)ldst, 0, 0);
#else
        asm volatile("global_load_async_to_lds_b128 %0, %1, off"
                     :: "v"((unsigned)(unsigned long long)ldst), "v"(gsrc)
                     : "memory");
#endif
    }
#if __has_builtin(__builtin_amdgcn_s_wait_asynccnt)
    __builtin_amdgcn_s_wait_asynccnt(0);
#else
    asm volatile("s_wait_asynccnt 0x0" ::: "memory");
#endif
    __syncthreads();

    v8f acc[8] = {};                      // 8 column tiles of 16x16 f32

    const float* arow = &vtile[n * LDS_ROW_STRIDE];   // this lane's A row (in LDS)
    const int ct0 = wave * 8;             // first column tile of this wave

    for (int kc = 0; kc < 32; ++kc) {
        // ---- build A fragment from LDS (f32 -> bf16) ----
        const float* p0 = arow + kc * 32 + g * 8;   // K run 0 (8 floats)
        const float* p1 = p0 + 16;                  // K run 1 (8 floats)
        v16bf a;
        #pragma unroll
        for (int e = 0; e < 8; ++e) a[e]     = (__bf16)p0[e];
        #pragma unroll
        for (int e = 0; e < 8; ++e) a[e + 8] = (__bf16)p1[e];

        // ---- 8 column tiles, B fragments stream from L2-resident packed W1
        #pragma unroll
        for (int ctl = 0; ctl < 8; ++ctl) {
            const int ct = ct0 + ctl;
            const v16bf bfrag =
                *(const v16bf*)(pw + (size_t)(((ct * 32 + kc) * 32) + lane) * 16);
            acc[ctl] = __builtin_amdgcn_wmma_f32_16x16x32_bf16(
                false, a, false, bfrag, (short)0, acc[ctl], false, false);
        }
    }

    // ---- fused epilogue: tanh(acc + pq + b1) * V, reduce over columns ----
    float rowpart[8];
    #pragma unroll
    for (int r = 0; r < 8; ++r) rowpart[r] = 0.0f;

    #pragma unroll
    for (int ctl = 0; ctl < 8; ++ctl) {
        const int col = (ct0 + ctl) * 16 + n;
        const float pqc = pq[b * U_ + col] + b1[col];
        const float vv  = Vv[col];
        #pragma unroll
        for (int r = 0; r < 8; ++r)
            rowpart[r] += tanhf(acc[ctl][r] + pqc) * vv;
    }

    // butterfly over the 16 lanes sharing a row group (masks stay in-half)
    #pragma unroll
    for (int r = 0; r < 8; ++r) {
        float v = rowpart[r];
        v += __shfl_xor(v, 1, 32);
        v += __shfl_xor(v, 2, 32);
        v += __shfl_xor(v, 4, 32);
        v += __shfl_xor(v, 8, 32);
        if (n == 0) atomicAdd(&rowsum[r + g * 8], v);   // ds_add_f32 across waves
    }
    __syncthreads();

    if (tid < 16) {
        const int t = t0 + tid;
        scores[b * T_ + t] = rowsum[tid] + bV[0] + mask[b * T_ + t] * (-1e9f);
    }
}

// ---------------------------------------------------------------------------
// Kernel 4: softmax over T per batch -> attention weights
// ---------------------------------------------------------------------------
__global__ __launch_bounds__(256) void softmax_kernel(const float* __restrict__ scores,
                                                      float* __restrict__ attn) {
    const int b   = blockIdx.x;
    const int tid = threadIdx.x;
    __shared__ float red[256];

    float m = -1e30f;
    for (int t = tid; t < T_; t += 256) m = fmaxf(m, scores[b * T_ + t]);
    red[tid] = m; __syncthreads();
    for (int s = 128; s > 0; s >>= 1) {
        if (tid < s) red[tid] = fmaxf(red[tid], red[tid + s]);
        __syncthreads();
    }
    m = red[0]; __syncthreads();

    float sum = 0.0f;
    for (int t = tid; t < T_; t += 256) sum += __expf(scores[b * T_ + t] - m);
    red[tid] = sum; __syncthreads();
    for (int s = 128; s > 0; s >>= 1) {
        if (tid < s) red[tid] += red[tid + s];
        __syncthreads();
    }
    const float inv = 1.0f / red[0];

    for (int t = tid; t < T_; t += 256)
        attn[b * T_ + t] = __expf(scores[b * T_ + t] - m) * inv;
}

// ---------------------------------------------------------------------------
// Kernel 5/6: context[b,d] = sum_t attn[b,t] * values[b,t,d]
// Deterministic two-stage reduction (no float atomics -> graph-replay stable).
// ---------------------------------------------------------------------------
__global__ __launch_bounds__(256) void context_partial_kernel(const float* __restrict__ values,
                                                              const float* __restrict__ attn,
                                                              float* __restrict__ partial) {
    const int b   = blockIdx.x;
    const int tc  = blockIdx.y;          // 0..15, chunk of 128 t's
    const int d   = threadIdx.x * 4;
    float4 acc = {0.f, 0.f, 0.f, 0.f};
    for (int t = tc * 128; t < tc * 128 + 128; ++t) {
        const float w = attn[b * T_ + t];
        const float4 v = *(const float4*)(values + ((size_t)b * T_ + t) * (size_t)D_ + d);
        acc.x += w * v.x; acc.y += w * v.y; acc.z += w * v.z; acc.w += w * v.w;
    }
    *(float4*)(partial + (size_t)(b * 16 + tc) * D_ + d) = acc;
}

__global__ __launch_bounds__(256) void context_final_kernel(const float* __restrict__ partial,
                                                            float* __restrict__ ctx) {
    const int idx = blockIdx.x * 256 + threadIdx.x;   // 0 .. 32767
    const int b = idx >> 10, d = idx & 1023;
    float s = 0.0f;
    #pragma unroll
    for (int tc = 0; tc < 16; ++tc) s += partial[(size_t)(b * 16 + tc) * D_ + d];
    ctx[idx] = s;
}

// ---------------------------------------------------------------------------
// Launch
// Inputs:  0 values[B,T,D]  1 query[B,D]  2 mask[B,T]
//          3 W1[D,U] 4 b1[U] 5 W2[D,U] 6 b2[U] 7 V[U,1] 8 bV[1]
// Output:  d_out = [ context[B*D] | attention_weights[B*T] ]  (f32)
// ---------------------------------------------------------------------------
extern "C" void kernel_launch(void* const* d_in, const int* in_sizes, int n_in,
                              void* d_out, int out_size, void* d_ws, size_t ws_size,
                              hipStream_t stream) {
    const float* values = (const float*)d_in[0];
    const float* query  = (const float*)d_in[1];
    const float* maskp  = (const float*)d_in[2];
    const float* W1     = (const float*)d_in[3];
    const float* b1     = (const float*)d_in[4];
    const float* W2     = (const float*)d_in[5];
    const float* b2     = (const float*)d_in[6];
    const float* Vv     = (const float*)d_in[7];
    const float* bV     = (const float*)d_in[8];

    float* ctx_out  = (float*)d_out;                       // [B*D]
    float* attn_out = (float*)d_out + (B_ * D_);           // [B*T]

    // Workspace layout (bytes): pw 2 MiB | pq 128 KiB | scores 256 KiB | partial 2 MiB
    char* ws = (char*)d_ws;
    __bf16* pw      = (__bf16*)ws;                               // 1048576 * 2
    float*  pq      = (float*)(ws + (size_t)2 * 1024 * 1024);    // 32768 * 4
    float*  scores  = pq + B_ * U_;                              // 65536 * 4
    float*  partial = scores + B_ * T_;                          // 524288 * 4

    pack_w1_kernel<<<(64 * 32 * 32 * 16) / 256, 256, 0, stream>>>(W1, pw);
    proj_q_kernel<<<(B_ * U_) / 256, 256, 0, stream>>>(query, W2, b2, pq);
    score_kernel<<<B_ * (T_ / 16), 256, 0, stream>>>(values, pw, pq, b1, Vv, bV,
                                                     maskp, scores);
    softmax_kernel<<<B_, 256, 0, stream>>>(scores, attn_out);
    context_partial_kernel<<<dim3(B_, 16), 256, 0, stream>>>(values, attn_out, partial);
    context_final_kernel<<<(B_ * D_) / 256, 256, 0, stream>>>(partial, ctx_out);
}
